// Decoder_66168266162932
// MI455X (gfx1250) — compile-verified, matrix operand order checked by reading, not run on hardware
//
#include <hip/hip_runtime.h>
#include <hip/hip_bf16.h>
#include <cstdint>
#include <cstddef>

// ---------------------------------------------------------------------------
// Attention-LSTM decoder for gfx1250 (MI455X).
// GEMMs: v_wmma_f32_16x16x32_bf16 (fp32->bf16 in registers, fp32 accum).
// B tiles double-buffered in LDS and shared by 4 waves (4 M-tiles) per block;
// staged with GLOBAL_LOAD_ASYNC_TO_LDS (ASYNCcnt) when the toolchain exposes
// the builtin, else global_load+ds_store. Weights (~70MB) stay L2-resident
// (192MB L2), so the 160 sequential steps re-read them from L2, not HBM.
// ---------------------------------------------------------------------------

typedef __attribute__((ext_vector_type(16))) __bf16 v16bf;
typedef __attribute__((ext_vector_type(8)))  float  v8f;
typedef __attribute__((ext_vector_type(4)))  int    v4i;

#define VOCABN 10000
#define EMBN   512
#define HIDN   1024
#define ENCN   2048
#define ATTN   512
#define PN     196
#define BN     64
#define TN     160
#define PADTOK 3

#if defined(__AMDGCN__) && __has_builtin(__builtin_amdgcn_global_load_async_to_lds_b128) && \
    __has_builtin(__builtin_amdgcn_s_wait_asynccnt)
#define USE_ASYNC_LDS 1
// builtin signature (from hipcc diagnostic): param0 = AS1 int4*, then LDS ptr,
// imm offset, imm cpol.
#define ASYNC_B128(gp, lp)                                                    \
    __builtin_amdgcn_global_load_async_to_lds_b128(                           \
        (__attribute__((address_space(1))) v4i*)(gp),                         \
        (__attribute__((address_space(3))) v4i*)(lp), 0, 0)
#else
#define USE_ASYNC_LDS 0
#endif

#define BSTRIDE 36   // padded LDS row stride (floats): 16B aligned, kills bank conflicts

// ---------------------------------------------------------------------------
// WMMA GEMM:  C[M,N] (+)= A[M,K] * W[N,K]^T  (+ bias[N])
// Row-major A (lda), row-major W ([out,in] as in the reference), row-major C.
// Optional A-row permutation: a_row = perm[row/permGroup]*permGroup + row%permGroup.
//
// Block = 4 waves (128 thr). Block owns N tile n0 (16 cols) and M slab of 64
// rows; wave w computes M tile m0 = blockIdx.y*64 + w*16. The B tile
// (16 x 32 fp32) is shared by all 4 waves through LDS, double buffered, and
// staged one k-step ahead (async when available). Requires M%64==0, N%16==0,
// K%32==0 -- true for every call site here.
//
// bf16 A-fragment lane layout (CDNA5 ISA 7.12.2, 16-bit A 16x32):
//   lanes 0-15 : M = lane,    K = {k0..k0+7} then {k0+16..k0+23}
//   lanes 16-31: M = lane-16, K = {k0+8..k0+15} then {k0+24..k0+31}
// B (32x16, KxN) uses the symmetric layout with N in place of M.
// C/D f32: lanes 0-15 N=lane M=g; lanes 16-31 N=lane-16 M=8+g (g = vgpr idx).
// ---------------------------------------------------------------------------
__global__ void __launch_bounds__(128)
wmma_gemm_nt(const float* __restrict__ A, int lda,
             const float* __restrict__ W, int ldw,
             const float* __restrict__ bias,
             float* __restrict__ C, int ldc,
             int M, int N, int K,
             const int* __restrict__ perm, int permGroup,
             int accum)
{
    __shared__ float Bs[2][16 * BSTRIDE];

    const int lane = threadIdx.x;            // 0..31 (wave32)
    const int wv   = threadIdx.y;            // 0..3 -> M tile within block
    const int tid  = wv * 32 + lane;         // 0..127
    const int n0   = blockIdx.x * 16;
    const int m0   = blockIdx.y * 64 + wv * 16;

    const int half = lane >> 4;              // 0 or 1
    const int lr   = lane & 15;

    int arow = m0 + lr;
    if (perm) arow = perm[arow / permGroup] * permGroup + (arow % permGroup);
    const float* __restrict__ Arow = A + (size_t)arow * lda;

    // staging map: thread tid copies one float4: row = tid/8, chunk = (tid%8)*4
    const int srow = tid >> 3;
    const int scol = (tid & 7) * 4;
    const float* __restrict__ Wstage = W + (size_t)(n0 + srow) * ldw + scol;

    v8f acc;
    if (accum) {
        #pragma unroll
        for (int g = 0; g < 8; ++g)
            acc[g] = C[(size_t)(m0 + half * 8 + g) * ldc + (n0 + lr)];
    } else {
        #pragma unroll
        for (int g = 0; g < 8; ++g) acc[g] = 0.0f;
    }

    const int nk = K >> 5;                   // K / 32

    // prologue: stage k-step 0 into buffer 0
    {
        float* lp = &Bs[0][srow * BSTRIDE + scol];
        float* gp = const_cast<float*>(Wstage);   // k0 = 0
#if USE_ASYNC_LDS
        ASYNC_B128(gp, lp);
#else
        *reinterpret_cast<float4*>(lp) = *reinterpret_cast<const float4*>(gp);
#endif
    }

    #pragma unroll 2
    for (int kt = 0; kt < nk; ++kt) {
        const int cur = kt & 1;
#if USE_ASYNC_LDS
        __builtin_amdgcn_s_wait_asynccnt(0);
#endif
        __syncthreads();                     // buffer `cur` ready for all waves

        if (kt + 1 < nk) {                   // stage next k-step into other buffer
            float* lp = &Bs[1 - cur][srow * BSTRIDE + scol];
            float* gp = const_cast<float*>(Wstage + (size_t)(kt + 1) * 32);
#if USE_ASYNC_LDS
            ASYNC_B128(gp, lp);
#else
            *reinterpret_cast<float4*>(lp) = *reinterpret_cast<const float4*>(gp);
#endif
        }

        // ---- A fragment: direct from global (per-wave M tile), fp32->bf16
        const int k0 = kt * 32;
        const int ka = k0 + half * 8;
        const int kb = k0 + 16 + half * 8;
        float4 a0 = *reinterpret_cast<const float4*>(Arow + ka);
        float4 a1 = *reinterpret_cast<const float4*>(Arow + ka + 4);
        float4 a2 = *reinterpret_cast<const float4*>(Arow + kb);
        float4 a3 = *reinterpret_cast<const float4*>(Arow + kb + 4);
        if (kt + 1 < nk)                     // global_prefetch_b8 for next A slab
            __builtin_prefetch(Arow + k0 + 32, 0, 3);

        // ---- B fragment: from LDS (shared by the 4 waves), fp32->bf16
        const float* bs = &Bs[cur][lr * BSTRIDE];
        float4 b0 = *reinterpret_cast<const float4*>(bs + half * 8);
        float4 b1 = *reinterpret_cast<const float4*>(bs + half * 8 + 4);
        float4 b2 = *reinterpret_cast<const float4*>(bs + 16 + half * 8);
        float4 b3 = *reinterpret_cast<const float4*>(bs + 16 + half * 8 + 4);

        const float av[16] = {a0.x,a0.y,a0.z,a0.w, a1.x,a1.y,a1.z,a1.w,
                              a2.x,a2.y,a2.z,a2.w, a3.x,a3.y,a3.z,a3.w};
        const float bv[16] = {b0.x,b0.y,b0.z,b0.w, b1.x,b1.y,b1.z,b1.w,
                              b2.x,b2.y,b2.z,b2.w, b3.x,b3.y,b3.z,b3.w};
        v16bf afrag, bfrag;
        #pragma unroll
        for (int i = 0; i < 16; ++i) {
            afrag[i] = (__bf16)av[i];
            bfrag[i] = (__bf16)bv[i];
        }
        acc = __builtin_amdgcn_wmma_f32_16x16x32_bf16(
                  false, afrag, false, bfrag, (short)0, acc, false, false);
    }

    const float bcol = bias ? bias[n0 + lr] : 0.0f;
    #pragma unroll
    for (int g = 0; g < 8; ++g)
        C[(size_t)(m0 + half * 8 + g) * ldc + (n0 + lr)] = acc[g] + bcol;
}

// ---------------------------------------------------------------------------
// Stable argsort by descending length (matches jnp.argsort(-lens)).
// idx[rank] = original index; inv[i] = rank of original i (so idx[inv[r]]=r).
// ---------------------------------------------------------------------------
__global__ void k_sort(const int* __restrict__ lens, int* __restrict__ idx,
                       int* __restrict__ inv, int* __restrict__ lens_s)
{
    __shared__ int sl[BN];
    int i = threadIdx.x;
    sl[i] = lens[i];
    __syncthreads();
    int li = sl[i];
    int rank = 0;
    for (int j = 0; j < BN; ++j) {
        int lj = sl[j];
        if (lj > li || (lj == li && j < i)) rank++;
    }
    idx[rank]    = i;
    inv[i]       = rank;
    lens_s[rank] = li;
}

// h0/c0 = concat(enc_h[0], enc_h[1], axis=1) gathered into sorted order.
__global__ void k_init_state(const float* __restrict__ enc_h,
                             const float* __restrict__ enc_c,
                             const int*   __restrict__ idx,
                             float* __restrict__ h, float* __restrict__ c)
{
    int i = blockIdx.x * blockDim.x + threadIdx.x;   // over B*HID
    if (i >= BN * HIDN) return;
    int r = i / HIDN, j = i % HIDN;
    int src = idx[r];
    const int hh = HIDN / 2;
    float vh, vc;
    if (j < hh) {
        vh = enc_h[(size_t)src * hh + j];
        vc = enc_c[(size_t)src * hh + j];
    } else {
        vh = enc_h[(size_t)(BN + src) * hh + (j - hh)];
        vc = enc_c[(size_t)(BN + src) * hh + (j - hh)];
    }
    h[i] = vh;
    c[i] = vc;
}

// e = relu(att1 + att2) . wfa + bfa ; softmax over P ; emit alpha (+ output).
__global__ void __launch_bounds__(256)
k_att_softmax(const float* __restrict__ att1,   // [B*P, ATT] (sorted rows)
              const float* __restrict__ att2,   // [B, ATT]
              const float* __restrict__ wfa,
              const float* __restrict__ bfa,    // scalar
              const int*   __restrict__ lens_s,
              const int*   __restrict__ inv,
              float* __restrict__ alpha,        // [B,P] workspace
              float* __restrict__ out_alphas,   // [B,T,P] final (ref ordering)
              int t)
{
    __shared__ float esh[PN];
    __shared__ float s_max, s_sum;
    const int b    = blockIdx.x;
    const int wave = threadIdx.x >> 5;
    const int lane = threadIdx.x & 31;
    const float* __restrict__ a2 = att2 + (size_t)b * ATTN;

    for (int p = wave; p < PN; p += 8) {
        const float* __restrict__ a1 = att1 + ((size_t)b * PN + p) * ATTN;
        float s = 0.0f;
        for (int a = lane; a < ATTN; a += 32) {
            float v = a1[a] + a2[a];
            v = v > 0.0f ? v : 0.0f;
            s += v * wfa[a];
        }
        #pragma unroll
        for (int m = 16; m >= 1; m >>= 1) s += __shfl_xor(s, m, 32);
        if (lane == 0) esh[p] = s + bfa[0];
    }
    __syncthreads();
    if (threadIdx.x == 0) {
        float mx = -3.4e38f;
        for (int p = 0; p < PN; ++p) mx = fmaxf(mx, esh[p]);
        s_max = mx;
    }
    __syncthreads();
    for (int p = threadIdx.x; p < PN; p += blockDim.x)
        esh[p] = __expf(esh[p] - s_max);
    __syncthreads();
    if (threadIdx.x == 0) {
        float s = 0.0f;
        for (int p = 0; p < PN; ++p) s += esh[p];
        s_sum = s;
    }
    __syncthreads();
    const int active = (t < lens_s[b]);
    const int orow   = inv[b];                  // replicate outputs[idx] reindex
    for (int p = threadIdx.x; p < PN; p += blockDim.x) {
        float al = esh[p] / s_sum;
        alpha[(size_t)b * PN + p] = al;
        out_alphas[((size_t)orow * TN + t) * PN + p] = active ? al : 0.0f;
    }
}

// context[b,e] = sum_p alpha[b,p] * img[idx[b],p,e]
__global__ void k_context(const float* __restrict__ img,
                          const float* __restrict__ alpha,
                          const int*   __restrict__ idx,
                          float* __restrict__ ctx)
{
    int b = blockIdx.y;
    int e = blockIdx.x * blockDim.x + threadIdx.x;
    const float* __restrict__ im = img + ((size_t)idx[b] * PN) * ENCN + e;
    const float* __restrict__ al = alpha + (size_t)b * PN;
    float s = 0.0f;
    for (int p = 0; p < PN; ++p) s += al[p] * im[(size_t)p * ENCN];
    ctx[(size_t)b * ENCN + e] = s;
}

// inp = concat(emb[token], sigmoid(gatepre) * context)
__global__ void k_build_inp(const int*   __restrict__ sent,
                            const int*   __restrict__ idx,
                            const float* __restrict__ emb,
                            const float* __restrict__ gatepre,
                            const float* __restrict__ ctx,
                            float* __restrict__ inp, int t)
{
    const int W = EMBN + ENCN;
    int i = blockIdx.x * blockDim.x + threadIdx.x;
    if (i >= BN * W) return;
    int b = i / W, j = i % W;
    float v;
    if (j < EMBN) {
        int tok = sent[(size_t)idx[b] * TN + t];
        v = (tok == PADTOK) ? 0.0f : emb[(size_t)tok * EMBN + j];
    } else {
        int j2 = j - EMBN;
        float g = gatepre[(size_t)b * ENCN + j2];
        g = 1.0f / (1.0f + __expf(-g));
        v = g * ctx[(size_t)b * ENCN + j2];
    }
    inp[i] = v;
}

// LSTM pointwise + ragged masking + masked output write (sorted -> inv row).
__global__ void k_lstm(const float* __restrict__ gates,
                       const int*   __restrict__ lens_s,
                       const int*   __restrict__ inv,
                       float* __restrict__ h, float* __restrict__ c,
                       float* __restrict__ out_h, int t)
{
    int i = blockIdx.x * blockDim.x + threadIdx.x;   // B*HID
    if (i >= BN * HIDN) return;
    int b = i / HIDN, j = i % HIDN;
    const float* __restrict__ g = gates + (size_t)b * 4 * HIDN;
    float gi = g[j];
    float gf = g[HIDN + j];
    float gg = g[2 * HIDN + j];
    float go = g[3 * HIDN + j];
    float si = 1.0f / (1.0f + __expf(-gi));
    float sf = 1.0f / (1.0f + __expf(-gf));
    float so = 1.0f / (1.0f + __expf(-go));
    float cn = sf * c[i] + si * tanhf(gg);
    float hn = so * tanhf(cn);
    int active = (t < lens_s[b]);
    if (active) { h[i] = hn; c[i] = cn; }
    out_h[((size_t)inv[b] * TN + t) * HIDN + j] = active ? hn : 0.0f;
}

// ---------------------------------------------------------------------------
extern "C" void kernel_launch(void* const* d_in, const int* in_sizes, int n_in,
                              void* d_out, int out_size, void* d_ws, size_t ws_size,
                              hipStream_t stream)
{
    (void)in_sizes; (void)n_in; (void)out_size; (void)ws_size;

    const float* img  = (const float*)d_in[0];   // [B,P,ENC]
    const int*   sent = (const int*)  d_in[1];   // [B,T]
    const float* ench = (const float*)d_in[2];   // [2,B,HID/2]
    const float* encc = (const float*)d_in[3];   // [2,B,HID/2]
    const int*   lens = (const int*)  d_in[4];   // [B]
    const float* emb  = (const float*)d_in[5];   // [VOCAB,EMB]
    const float* Wea  = (const float*)d_in[6];   // [ATT,ENC]
    const float* bea  = (const float*)d_in[7];
    const float* Wda  = (const float*)d_in[8];   // [ATT,HID]
    const float* bda  = (const float*)d_in[9];
    const float* wfa  = (const float*)d_in[10];  // [ATT]
    const float* bfa  = (const float*)d_in[11];  // scalar
    const float* Wfb  = (const float*)d_in[12];  // [ENC,HID]
    const float* bfb  = (const float*)d_in[13];
    const float* Wih  = (const float*)d_in[14];  // [4H, EMB+ENC]
    const float* bih  = (const float*)d_in[15];
    const float* Whh  = (const float*)d_in[16];  // [4H, HID]
    const float* bhh  = (const float*)d_in[17];

    float* out_outputs = (float*)d_out;                              // [B,T,HID]
    float* out_alphas  = (float*)d_out + (size_t)BN * TN * HIDN;     // [B,T,P]

    // ---- workspace carve-up (256B aligned slices) ----
    char* ws = (char*)d_ws;
    size_t off = 0;
    auto carve = [&](size_t bytes) -> void* {
        void* p = ws + off;
        off += (bytes + 255) & ~(size_t)255;
        return p;
    };
    int*   idx     = (int*)  carve(BN * sizeof(int));
    int*   inv     = (int*)  carve(BN * sizeof(int));
    int*   lens_s  = (int*)  carve(BN * sizeof(int));
    float* h       = (float*)carve((size_t)BN * HIDN * sizeof(float));
    float* c       = (float*)carve((size_t)BN * HIDN * sizeof(float));
    float* att1    = (float*)carve((size_t)BN * PN * ATTN * sizeof(float));
    float* att2    = (float*)carve((size_t)BN * ATTN * sizeof(float));
    float* alpha   = (float*)carve((size_t)BN * PN * sizeof(float));
    float* ctx     = (float*)carve((size_t)BN * ENCN * sizeof(float));
    float* gatepre = (float*)carve((size_t)BN * ENCN * sizeof(float));
    float* inp     = (float*)carve((size_t)BN * (EMBN + ENCN) * sizeof(float));
    float* gates   = (float*)carve((size_t)BN * 4 * HIDN * sizeof(float));

    const dim3 wblk(32, 4, 1);

    // 1) sort by descending length
    k_sort<<<1, BN, 0, stream>>>(lens, idx, inv, lens_s);

    // 2) gather initial h0/c0 in sorted order
    k_init_state<<<(BN * HIDN + 255) / 256, 256, 0, stream>>>(ench, encc, idx, h, c);

    // 3) hoisted encoder projection: att1 = img[idx] @ Wea^T + bea  [B*P, ATT]
    wmma_gemm_nt<<<dim3(ATTN / 16, (BN * PN) / 64), wblk, 0, stream>>>(
        img, ENCN, Wea, ENCN, bea, att1, ATTN,
        BN * PN, ATTN, ENCN, idx, PN, 0);

    // 4) sequential scan over T
    for (int t = 0; t < TN; ++t) {
        // att2 = h @ Wda^T + bda        [B, ATT]
        wmma_gemm_nt<<<dim3(ATTN / 16, BN / 64), wblk, 0, stream>>>(
            h, HIDN, Wda, HIDN, bda, att2, ATTN,
            BN, ATTN, HIDN, nullptr, 1, 0);

        // e / softmax / alpha (+ masked alpha output)
        k_att_softmax<<<BN, 256, 0, stream>>>(
            att1, att2, wfa, bfa, lens_s, inv, alpha, out_alphas, t);

        // context = alpha . img[idx]
        k_context<<<dim3(ENCN / 256, BN), 256, 0, stream>>>(img, alpha, idx, ctx);

        // gate pre-activation: h @ Wfb^T + bfb   [B, ENC]
        wmma_gemm_nt<<<dim3(ENCN / 16, BN / 64), wblk, 0, stream>>>(
            h, HIDN, Wfb, HIDN, bfb, gatepre, ENCN,
            BN, ENCN, HIDN, nullptr, 1, 0);

        // inp = [x_t, sigmoid(gatepre) * context]
        k_build_inp<<<(BN * (EMBN + ENCN) + 255) / 256, 256, 0, stream>>>(
            sent, idx, emb, gatepre, ctx, inp, t);

        // gates = inp @ Wih^T + bih
        wmma_gemm_nt<<<dim3((4 * HIDN) / 16, BN / 64), wblk, 0, stream>>>(
            inp, EMBN + ENCN, Wih, EMBN + ENCN, bih, gates, 4 * HIDN,
            BN, 4 * HIDN, EMBN + ENCN, nullptr, 1, 0);

        // gates += h @ Whh^T + bhh
        wmma_gemm_nt<<<dim3((4 * HIDN) / 16, BN / 64), wblk, 0, stream>>>(
            h, HIDN, Whh, HIDN, bhh, gates, 4 * HIDN,
            BN, 4 * HIDN, HIDN, nullptr, 1, 1);

        // LSTM pointwise + ragged masking + output write
        k_lstm<<<(BN * HIDN + 255) / 256, 256, 0, stream>>>(
            gates, lens_s, inv, h, c, out_outputs, t);
    }
}
